// SequentialDecoder_49555332661836
// MI455X (gfx1250) — matrix operand order repaired and checked by reading, not compiled, not run on hardware
//
#include <hip/hip_runtime.h>
#include <hip/hip_bf16.h>

typedef _Float16 h16;
typedef __attribute__((ext_vector_type(8)))  _Float16 v8h;
typedef __attribute__((ext_vector_type(16))) _Float16 v16h;
typedef __attribute__((ext_vector_type(8)))  float    v8f;

#define HS  520   // s_hin row stride (halves), 512 + 8 pad
#define XS  1032  // s_xc/s_sz row stride (halves), 1024 + 8 pad
#define DTS 40    // s_dtr row stride (halves), 32 + 8 pad
#define MS  520   // s_mh row stride (halves), 512 + 8 pad

// ---------------- fast math helpers (v_rcp/v_exp/v_log, no IEEE div) ----------------
__device__ __forceinline__ float fast_rcp(float x) { return __builtin_amdgcn_rcpf(x); }
__device__ __forceinline__ float siluf(float x) {
  return x * fast_rcp(1.0f + __expf(-x));
}
__device__ __forceinline__ float softplusf(float x) {
  return (x > 15.0f) ? x : __logf(1.0f + __expf(x));
}

// A fragment (16x32 f16 tile, row-major in LDS).
// Contiguous k order used here (same as the B pack): lanes 0-15 row M=lane, K=0..15; lanes 16-31 row M=lane-16, K=16..31.
__device__ __forceinline__ v16h frag_a_lds(const h16* tile, int stride, int kt, int lane) {
  int row = lane & 15;
  int kb  = kt * 32 + ((lane & 16) ? 16 : 0);
  const h16* p = tile + row * stride + kb;
  v8h lo = *(const v8h*)(p);
  v8h hi = *(const v8h*)(p + 8);
  return __builtin_shufflevector(lo, hi, 0,1,2,3,4,5,6,7,8,9,10,11,12,13,14,15);
}

// B fragment from packed weights: frag (kt,nt) stores 32 lanes x 16 contiguous halves.
// lane<16: K = kt*32 + j; lane>=16: K = kt*32+16+j; N = nt*16 + (lane&15).
__device__ __forceinline__ v16h frag_b_glb(const h16* pack, int nTiles, int kt, int nt, int lane) {
  const h16* p = pack + (((size_t)kt * nTiles + nt) * 32 + (size_t)lane) * 16;
  return *(const v16h*)p;
}

__device__ __forceinline__ v8f wmma16(v16h a, v16h b, v8f c) {
  return __builtin_amdgcn_wmma_f32_16x16x32_f16(false, a, false, b, (short)0, c, false, false);
}

// ---------------- preprocessing kernels ----------------

// fused = layernorm(relu([tc,cc] @ W_cf + b_cf)) ; one block per row
__global__ void fused_kernel(const float* __restrict__ tc, const float* __restrict__ cc,
                             const float* __restrict__ W_cf, const float* __restrict__ b_cf,
                             const float* __restrict__ ln_g, const float* __restrict__ ln_b,
                             float* __restrict__ fused) {
  __shared__ float rowbuf[1024];
  __shared__ float hbuf[512];
  __shared__ float red[256], red2[256];
  int row = blockIdx.x, tid = threadIdx.x;
  for (int i = tid; i < 512; i += 256) {
    rowbuf[i]       = tc[row * 512 + i];
    rowbuf[512 + i] = cc[row * 512 + i];
  }
  __syncthreads();
  float h0 = 0.f, h1 = 0.f;
  {
    int c0 = tid, c1 = tid + 256;
    float a0 = 0.f, a1 = 0.f;
    for (int k = 0; k < 1024; ++k) {
      float r = rowbuf[k];
      a0 += r * W_cf[k * 512 + c0];
      a1 += r * W_cf[k * 512 + c1];
    }
    h0 = fmaxf(a0 + b_cf[c0], 0.f);
    h1 = fmaxf(a1 + b_cf[c1], 0.f);
    hbuf[c0] = h0; hbuf[c1] = h1;
  }
  red[tid] = h0 + h1;
  red2[tid] = h0 * h0 + h1 * h1;
  __syncthreads();
  for (int o = 128; o > 0; o >>= 1) {
    if (tid < o) { red[tid] += red[tid + o]; red2[tid] += red2[tid + o]; }
    __syncthreads();
  }
  float mu = red[0] / 512.0f;
  float var = red2[0] / 512.0f - mu * mu;
  float rs = rsqrtf(var + 1e-5f);
  for (int c = tid; c < 512; c += 256)
    fused[row * 512 + c] = (hbuf[c] - mu) * rs * ln_g[c] + ln_b[c];
}

// hin_base = fused @ W_feat[:512] + b_feat ; one block per row
__global__ void hinbase_kernel(const float* __restrict__ fused, const float* __restrict__ W_feat,
                               const float* __restrict__ b_feat, float* __restrict__ hin_base) {
  __shared__ float fbuf[512];
  int row = blockIdx.x, tid = threadIdx.x;
  for (int i = tid; i < 512; i += 256) fbuf[i] = fused[row * 512 + i];
  __syncthreads();
  int c0 = tid, c1 = tid + 256;
  float a0 = b_feat[c0], a1 = b_feat[c1];
  for (int k = 0; k < 512; ++k) {
    float f = fbuf[k];
    a0 += f * W_feat[k * 512 + c0];
    a1 += f * W_feat[k * 512 + c1];
  }
  hin_base[row * 512 + c0] = a0;
  hin_base[row * 512 + c1] = a1;
}

// pack a (K,N) f32 row-major weight into fragment-major f16 WMMA B layout
__global__ void pack_kernel(const float* __restrict__ src, h16* __restrict__ dst, int K, int N) {
  int e = blockIdx.x * 256 + threadIdx.x;
  if (e >= K * N) return;
  int nTiles = N >> 4;
  int j = e & 15;
  int lane = (e >> 4) & 31;
  int frag = e >> 9;
  int nt = frag % nTiles;
  int kt = frag / nTiles;
  int k = kt * 32 + ((lane >> 4) << 4) + j;
  int n = nt * 16 + (lane & 15);
  dst[e] = (h16)src[(size_t)k * N + n];
}

// pack [W_tr | W_se | 0 0] (512 x 16, cols 0-6 trend, 7-13 seas) into one-ntile B layout
__global__ void pack_ts_kernel(const float* __restrict__ W_tr, const float* __restrict__ W_se,
                               h16* __restrict__ dst) {
  int e = blockIdx.x * 256 + threadIdx.x;  // 512*16 elements, nTiles = 1
  if (e >= 512 * 16) return;
  int j = e & 15;
  int lane = (e >> 4) & 31;
  int kt = e >> 9;
  int k = kt * 32 + ((lane >> 4) << 4) + j;
  int n = lane & 15;
  float v = 0.f;
  if (n < 7)       v = W_tr[k * 7 + n];
  else if (n < 14) v = W_se[k * 7 + (n - 7)];
  dst[e] = (h16)v;
}

// ---------------- main sequential decoder ----------------
// 64 blocks x 256 threads (8 waves); 16 rows per block; 64 steps internal.
__global__ void decoder_kernel(
    const float* __restrict__ hin_base, const float* __restrict__ W_feat,
    const h16* __restrict__ pW_in, const float* __restrict__ b_inproj,
    const float* __restrict__ conv_w, const float* __restrict__ conv_b,
    const h16* __restrict__ pW_x,
    const h16* __restrict__ pW_dt, const float* __restrict__ b_dt,
    const float* __restrict__ D_param,
    const h16* __restrict__ pW_out, const float* __restrict__ b_out,
    const h16* __restrict__ pW_ts,
    const float* __restrict__ b_tr, const float* __restrict__ b_se,
    const float* __restrict__ W_fin, const float* __restrict__ b_fin,
    const float* __restrict__ init_vals,
    float* __restrict__ out_tr, float* __restrict__ out_se, float* __restrict__ out_co) {
  extern __shared__ char smem[];
  // time-multiplexed regions:
  h16*   s_hin  = (h16*)smem;                    // 16*520*2 = 16640 B   (hin build + xz GEMM)
  float* s_proj = (float*)smem;                  // 16*160*4 = 10240 B   (aliases s_hin; proj..dt)
  h16*   s_dtr  = (h16*)(smem + 10240);          // 16*40*2  = 1280 B
  float* s_bc   = (float*)(smem + 11520);        // 64 B
  h16*   s_xc   = (h16*)(smem + 16640);          // 16*1032*2 = 33024 B
  h16*   s_mh   = (h16*)(smem + 16640);          // 16*520*2 = 16640 B   (aliases s_xc; after y)
  h16*   s_sz   = (h16*)(smem + 49664);          // 33024 B  (silu(z), later y in-place)
  float* s_ts   = (float*)(smem + 82688);        // 16*14*4 = 896 B
  float* s_comb = (float*)(smem + 83584);        // 16*7*4 = 448 B       total 84032 B

  const int tid = threadIdx.x;
  const int lane = tid & 31;
  const int wave = tid >> 5;
  const int rowBase = blockIdx.x * 16;
  const int col16 = lane & 15;
  const int rowOff = (lane & 16) ? 8 : 0;

  if (tid < 112) {
    int r = tid / 7, c = tid % 7;
    s_comb[r * 7 + c] = init_vals[(rowBase + r) * 7 + c];
  }
  __syncthreads();

  for (int t = 0; t < 64; ++t) {
    // ---- Stage 1: hin = hin_base + comb @ W_feat[512:519]  (f32 math -> f16 LDS)
    for (int e = tid; e < 16 * 512; e += 256) {
      int r = e >> 9, d = e & 511;
      float acc = hin_base[(size_t)(rowBase + r) * 512 + d];
#pragma unroll
      for (int c = 0; c < 7; ++c) acc += s_comb[r * 7 + c] * W_feat[(512 + c) * 512 + d];
      s_hin[r * HS + d] = (h16)acc;
    }
    __syncthreads();

    // ---- Stage 2: xz = hin @ W_inproj (+bias) -> x_c = silu(x*cw+cb), sz = silu(z)
    // K=512 (16 ktiles), N=2048 (128 ntiles); wave handles 16 ntiles in groups of 4.
#pragma unroll 1
    for (int g = 0; g < 4; ++g) {
      int nt0 = wave * 16 + g * 4;
      v8f a0 = {}, a1 = {}, a2 = {}, a3 = {};
#pragma unroll 2
      for (int kt = 0; kt < 16; ++kt) {
        v16h a = frag_a_lds(s_hin, HS, kt, lane);
        a0 = wmma16(a, frag_b_glb(pW_in, 128, kt, nt0 + 0, lane), a0);
        a1 = wmma16(a, frag_b_glb(pW_in, 128, kt, nt0 + 1, lane), a1);
        a2 = wmma16(a, frag_b_glb(pW_in, 128, kt, nt0 + 2, lane), a2);
        a3 = wmma16(a, frag_b_glb(pW_in, 128, kt, nt0 + 3, lane), a3);
      }
#pragma unroll
      for (int i = 0; i < 4; ++i) {
        v8f acc = (i == 0) ? a0 : (i == 1) ? a1 : (i == 2) ? a2 : a3;
        int n = (nt0 + i) * 16 + col16;
        float bias = b_inproj[n];
        if (n < 1024) {
          float cw = conv_w[n * 4 + 3], cb = conv_b[n];
#pragma unroll
          for (int v = 0; v < 8; ++v) {
            float x = acc[v] + bias;
            s_xc[(v + rowOff) * XS + n] = (h16)siluf(x * cw + cb);
          }
        } else {
          int nz = n - 1024;
#pragma unroll
          for (int v = 0; v < 8; ++v)
            s_sz[(v + rowOff) * XS + nz] = (h16)siluf(acc[v] + bias);
        }
      }
    }
    __syncthreads();

    // ---- Stage 3: proj = x_c @ W_xproj   K=1024 (32 kt), N=160 (10 nt)
    for (int nt = wave; nt < 10; nt += 8) {
      v8f acc = {};
#pragma unroll 2
      for (int kt = 0; kt < 32; ++kt) {
        v16h a = frag_a_lds(s_xc, XS, kt, lane);
        acc = wmma16(a, frag_b_glb(pW_x, 10, kt, nt, lane), acc);
      }
      int col = nt * 16 + col16;
#pragma unroll
      for (int v = 0; v < 8; ++v) s_proj[(v + rowOff) * 160 + col] = acc[v];
    }
    __syncthreads();

    // ---- Stage 4: bc[r] = dot(Bm, Cm); dt_r tile -> f16
    if (tid < 16) {
      const float* pr = s_proj + tid * 160;
      float acc = 0.f;
      for (int i = 0; i < 64; ++i) acc += pr[32 + i] * pr[96 + i];
      s_bc[tid] = acc;
    }
    for (int e = tid; e < 16 * 32; e += 256) {
      int r = e >> 5, k = e & 31;
      s_dtr[r * DTS + k] = (h16)s_proj[r * 160 + k];
    }
    __syncthreads();

    // ---- Stage 5: dt = softplus(dt_r @ W_dt + b_dt); y = (dt*bc + D)*x_c*silu(z) -> s_sz in place
    // K=32 (1 kt), N=1024 (64 nt); wave handles 8.
#pragma unroll 1
    for (int i = 0; i < 8; ++i) {
      int nt = wave * 8 + i;
      v16h a = frag_a_lds(s_dtr, DTS, 0, lane);
      v8f acc = {};
      acc = wmma16(a, frag_b_glb(pW_dt, 64, 0, nt, lane), acc);
      int col = nt * 16 + col16;
      float bd = b_dt[col], dp = D_param[col];
#pragma unroll
      for (int v = 0; v < 8; ++v) {
        int r = v + rowOff;
        float dt = softplusf(acc[v] + bd);
        float xc = (float)s_xc[r * XS + col];
        float sz = (float)s_sz[r * XS + col];
        s_sz[r * XS + col] = (h16)((dt * s_bc[r] + dp) * xc * sz);
      }
    }
    __syncthreads();

    // ---- Stage 6: m = y @ W_out + b_out (f16 into s_mh)   K=1024 (32 kt), N=512 (32 nt)
#pragma unroll 1
    for (int i = 0; i < 4; ++i) {
      int nt = wave * 4 + i;
      v8f acc = {};
#pragma unroll 2
      for (int kt = 0; kt < 32; ++kt) {
        v16h a = frag_a_lds(s_sz, XS, kt, lane);
        acc = wmma16(a, frag_b_glb(pW_out, 32, kt, nt, lane), acc);
      }
      int col = nt * 16 + col16;
      float bo = b_out[col];
#pragma unroll
      for (int v = 0; v < 8; ++v) s_mh[(v + rowOff) * MS + col] = (h16)(acc[v] + bo);
    }
    __syncthreads();

    // ---- Stage 7: [trend|seas] = m @ [W_tr|W_se]  as ONE 16-wide WMMA n-tile (wave 0 only)
    if (wave == 0) {
      v8f acc = {};
#pragma unroll 2
      for (int kt = 0; kt < 16; ++kt) {
        v16h a = frag_a_lds(s_mh, MS, kt, lane);
        acc = wmma16(a, frag_b_glb(pW_ts, 1, kt, 0, lane), acc);
      }
      int c = col16;
      if (c < 14) {
        float bb = (c < 7) ? b_tr[c] : b_se[c - 7];
#pragma unroll
        for (int v = 0; v < 8; ++v) s_ts[(v + rowOff) * 14 + c] = acc[v] + bb;
      }
    }
    __syncthreads();

    // ---- Stage 8: comb = [trend,seas] @ W_fin + b_fin; emit outputs (B,T,7)
    if (tid < 112) {
      int r = tid / 7, c = tid % 7;
      float acc = b_fin[c];
#pragma unroll
      for (int j = 0; j < 14; ++j) acc += s_ts[r * 14 + j] * W_fin[j * 7 + c];
      int grow = rowBase + r;
      size_t o = ((size_t)grow * 64 + t) * 7 + c;
      out_tr[o] = s_ts[r * 14 + c];
      out_se[o] = s_ts[r * 14 + 7 + c];
      out_co[o] = acc;
      s_comb[r * 7 + c] = acc;
    }
    __syncthreads();
  }
}

// ---------------- launch ----------------
extern "C" void kernel_launch(void* const* d_in, const int* in_sizes, int n_in,
                              void* d_out, int out_size, void* d_ws, size_t ws_size,
                              hipStream_t stream) {
  const float* tc       = (const float*)d_in[0];
  const float* cc       = (const float*)d_in[1];
  const float* init_v   = (const float*)d_in[2];
  const float* W_cf     = (const float*)d_in[3];
  const float* b_cf     = (const float*)d_in[4];
  const float* ln_g     = (const float*)d_in[5];
  const float* ln_b     = (const float*)d_in[6];
  const float* W_feat   = (const float*)d_in[7];
  const float* b_feat   = (const float*)d_in[8];
  const float* W_inproj = (const float*)d_in[9];
  const float* b_inproj = (const float*)d_in[10];
  const float* conv_w   = (const float*)d_in[11];
  const float* conv_b   = (const float*)d_in[12];
  const float* W_xproj  = (const float*)d_in[13];
  const float* W_dt     = (const float*)d_in[14];
  const float* b_dt     = (const float*)d_in[15];
  const float* D_param  = (const float*)d_in[16];
  const float* W_out    = (const float*)d_in[17];
  const float* b_out    = (const float*)d_in[18];
  const float* W_tr     = (const float*)d_in[19];
  const float* b_tr     = (const float*)d_in[20];
  const float* W_se     = (const float*)d_in[21];
  const float* b_se     = (const float*)d_in[22];
  const float* W_fin    = (const float*)d_in[23];
  const float* b_fin    = (const float*)d_in[24];

  // workspace layout
  float* fused    = (float*)d_ws;
  float* hin_base = fused + 1024 * 512;
  h16*   pW_in    = (h16*)(hin_base + 1024 * 512);
  h16*   pW_x     = pW_in + 512 * 2048;
  h16*   pW_dtm   = pW_x + 1024 * 160;
  h16*   pW_outm  = pW_dtm + 32 * 1024;
  h16*   pW_ts    = pW_outm + 1024 * 512;

  fused_kernel<<<1024, 256, 0, stream>>>(tc, cc, W_cf, b_cf, ln_g, ln_b, fused);
  hinbase_kernel<<<1024, 256, 0, stream>>>(fused, W_feat, b_feat, hin_base);
  pack_kernel<<<(512 * 2048 + 255) / 256, 256, 0, stream>>>(W_inproj, pW_in, 512, 2048);
  pack_kernel<<<(1024 * 160 + 255) / 256, 256, 0, stream>>>(W_xproj, pW_x, 1024, 160);
  pack_kernel<<<(32 * 1024 + 255) / 256, 256, 0, stream>>>(W_dt, pW_dtm, 32, 1024);
  pack_kernel<<<(1024 * 512 + 255) / 256, 256, 0, stream>>>(W_out, pW_outm, 1024, 512);
  pack_ts_kernel<<<(512 * 16 + 255) / 256, 256, 0, stream>>>(W_tr, W_se, pW_ts);

  float* out = (float*)d_out;
  float* out_tr = out;
  float* out_se = out + (size_t)1024 * 64 * 7;
  float* out_co = out + (size_t)2 * 1024 * 64 * 7;

  decoder_kernel<<<64, 256, 84032, stream>>>(
      hin_base, W_feat, pW_in, b_inproj, conv_w, conv_b, pW_x, pW_dtm, b_dt,
      D_param, pW_outm, b_out, pW_ts, b_tr, b_se, W_fin, b_fin, init_v,
      out_tr, out_se, out_co);
}
